// Sparsify_hypercol_local_modular_86337432584586
// MI455X (gfx1250) — compile-verified, matrix operand order checked by reading, not run on hardware
//
#include <hip/hip_runtime.h>
#include <hip/hip_bf16.h>
#include <stdint.h>

// Problem constants (from reference)
#define N_      128
#define CH_     64
#define RES_    64
#define LRF_    4
#define LH_     16
#define S_      256
#define B_      16
#define TOPK_   25
#define OC_     7
#define INDIM_  1856   // TOPK_*CH_ + S_
#define HID_    1856
#define OUTDIM_ 1792   // S_*OC_
#define ROWS_   2048   // N_*B_

// CDNA5 WMMA types (wave32)
typedef __attribute__((ext_vector_type(16))) __bf16 v16bf;
typedef __attribute__((ext_vector_type(8)))  float  v8f;

union FragU { v16bf v; uint4 q[2]; };   // 32 bytes: 16 bf16 per lane

__device__ __forceinline__ uint16_t f2bf(float f) {
  // round-to-nearest-even f32 -> bf16 (bit trick; NaN payload not preserved)
  uint32_t u = __float_as_uint(f);
  return (uint16_t)((u + 0x7FFFu + ((u >> 16) & 1u)) >> 16);
}

__device__ __forceinline__ void load_frag(FragU& f, const uint16_t* p, int ko) {
  f.q[0] = *(const uint4*)(p + ko);
  f.q[1] = *(const uint4*)(p + ko + 16);
}

// ---------------------------------------------------------------------------
// Kernel 1: W[k,n] (f32, row-major) -> Wt[n,k] (bf16, contraction-contiguous)
// ---------------------------------------------------------------------------
__global__ void k_transpose_f32_to_bf16(const float* __restrict__ W,
                                        uint16_t* __restrict__ Wt,
                                        int Kdim, int Ndim) {
  int g = blockIdx.x * blockDim.x + threadIdx.x;
  if (g >= Kdim * Ndim) return;
  int n = g / Kdim;
  int k = g - n * Kdim;
  Wt[(size_t)n * Kdim + k] = f2bf(W[(size_t)k * Ndim + n]);
}

// ---------------------------------------------------------------------------
// Kernel 2: per-(n,b) softmax + top-25 + bf16 feature-row assembly
// One 256-thread block per row; thread s owns spatial position s.
// ---------------------------------------------------------------------------
__global__ __launch_bounds__(256)
void k_topk_gather(const float* __restrict__ x,
                   const float* __restrict__ conv_w,
                   uint16_t* __restrict__ featB) {
  __shared__ float pS[S_];
  __shared__ float red[S_];
  __shared__ int   selS[S_];

  const int row = blockIdx.x;
  const int n = row / B_, b = row - n * B_;
  const int s = threadIdx.x;
  const int h = (b / LRF_) * LH_ + (s / LH_);
  const int w = (b % LRF_) * LH_ + (s % LH_);
  const float* xp = x + (size_t)n * CH_ * RES_ * RES_ + (size_t)h * RES_ + w;
  const float* cw = conv_w + b * CH_;

  // 1x1 conv logit: dot over 64 channels
  float t = 0.f;
  #pragma unroll 8
  for (int c = 0; c < CH_; ++c) t += xp[(size_t)c * RES_ * RES_] * cw[c];

  // softmax over the 256 positions
  red[s] = t; __syncthreads();
  for (int off = 128; off > 0; off >>= 1) {
    if (s < off) red[s] = fmaxf(red[s], red[s + off]);
    __syncthreads();
  }
  float mx = red[0]; __syncthreads();
  float e = __expf(t - mx);
  red[s] = e; __syncthreads();
  for (int off = 128; off > 0; off >>= 1) {
    if (s < off) red[s] += red[s + off];
    __syncthreads();
  }
  float p = e / red[0];
  pS[s] = p; __syncthreads();

  // exact top-K by rank (index tie-break == stable top_k), O(S^2) but tiny
  int rank = 0;
  for (int s2 = 0; s2 < S_; ++s2) {
    float q = pS[s2];
    rank += (q > p) || (q == p && s2 < s);
  }
  const int sel = (rank < TOPK_) ? 1 : 0;
  selS[s] = sel; __syncthreads();

  uint16_t* fr = featB + (size_t)row * INDIM_;
  fr[TOPK_ * CH_ + s] = f2bf((float)sel);   // straight-through mask == mask fwd

  if (sel) {
    int kidx = 0;                            // ascending-index compaction
    for (int s2 = 0; s2 < s; ++s2) kidx += selS[s2];
    uint16_t* dst = fr + kidx * CH_;         // position-major then channel
    #pragma unroll 8
    for (int c = 0; c < CH_; ++c) dst[c] = f2bf(xp[(size_t)c * RES_ * RES_]);
  }
}

// ---------------------------------------------------------------------------
// Kernels 3/4: bf16 WMMA GEMM  D[M,N] = act(A[M,K] @ Wt[N,K]^T + bias[N])
// 256 threads = 8 waves; block tile 128(M) x 64(N); waves 4(M) x 2(N);
// wave tile 32(M) x 32(N) = 4 accumulators, software-pipelined (register
// double buffer): fragments for K-step k+32 are in flight while the 4 WMMAs
// of step k execute, so waits become "allow in-flight next set" not 0.
// Fragment layout per ISA 16-bit A table: lane half (lane>>4) picks the
// 8-wide K sub-block; elements 0..7 -> K+0..7, 8..15 -> K+16..23 (two
// contiguous 16B chunks). B indexed by column (Wt is column-major of W).
// Requires Ktot % 64 == 0 (1856 = 29*64).
// ---------------------------------------------------------------------------
template <bool RELU, bool OUT_BF16>
__global__ __launch_bounds__(256)
void k_gemm_wmma(const uint16_t* __restrict__ A, int lda,
                 const uint16_t* __restrict__ Wt, int Ktot,
                 const float* __restrict__ bias,
                 void* __restrict__ out, int ldd) {
  const int lane  = threadIdx.x & 31;
  const int wid   = threadIdx.x >> 5;
  const int waveM = wid & 3;        // 4 waves along M
  const int waveN = wid >> 2;       // 2 waves along N
  const int half  = lane >> 4;
  const int lr    = lane & 15;
  const int m0 = blockIdx.y * 128 + waveM * 32;
  const int n0 = blockIdx.x * 64  + waveN * 32;

  const uint16_t* apr0 = A  + (size_t)(m0 + lr)      * lda;   // rows m0..m0+15
  const uint16_t* apr1 = A  + (size_t)(m0 + 16 + lr) * lda;   // rows m0+16..+31
  const uint16_t* bpr0 = Wt + (size_t)(n0 + lr)      * Ktot;  // cols n0..n0+15
  const uint16_t* bpr1 = Wt + (size_t)(n0 + 16 + lr) * Ktot;  // cols n0+16..+31

  v8f c00 = {}, c01 = {}, c10 = {}, c11 = {};
  FragU a0A, a1A, b0A, b1A;     // pipeline stage A (even K-steps)
  FragU a0B, a1B, b0B, b1B;     // pipeline stage B (odd K-steps)

  // prologue: stage A <- K-step 0
  {
    const int ko = half * 8;
    load_frag(a0A, apr0, ko); load_frag(a1A, apr1, ko);
    load_frag(b0A, bpr0, ko); load_frag(b1A, bpr1, ko);
  }

  for (int kk = 0; kk < Ktot; kk += 64) {
    // issue stage B loads (K-step kk+32; always valid, Ktot % 64 == 0)
    {
      const int ko = kk + 32 + half * 8;
      load_frag(a0B, apr0, ko); load_frag(a1B, apr1, ko);
      load_frag(b0B, bpr0, ko); load_frag(b1B, bpr1, ko);
    }
    if (kk + 128 <= Ktot) {               // look-ahead prefetch (2 steps out)
      const int ko = kk + 64 + half * 8;
      __builtin_prefetch(apr0 + ko, 0, 1);
      __builtin_prefetch(apr1 + ko, 0, 1);
      __builtin_prefetch(bpr0 + ko, 0, 1);
      __builtin_prefetch(bpr1 + ko, 0, 1);
    }
    // consume stage A (independent accumulators -> no D->A/B hazards)
    c00 = __builtin_amdgcn_wmma_f32_16x16x32_bf16(false, a0A.v, false, b0A.v,
                                                  (short)0, c00, false, false);
    c10 = __builtin_amdgcn_wmma_f32_16x16x32_bf16(false, a1A.v, false, b0A.v,
                                                  (short)0, c10, false, false);
    c01 = __builtin_amdgcn_wmma_f32_16x16x32_bf16(false, a0A.v, false, b1A.v,
                                                  (short)0, c01, false, false);
    c11 = __builtin_amdgcn_wmma_f32_16x16x32_bf16(false, a1A.v, false, b1A.v,
                                                  (short)0, c11, false, false);
    // issue stage A loads for the next 64-wide step (K-step kk+64)
    if (kk + 64 < Ktot) {
      const int ko = kk + 64 + half * 8;
      load_frag(a0A, apr0, ko); load_frag(a1A, apr1, ko);
      load_frag(b0A, bpr0, ko); load_frag(b1A, bpr1, ko);
    }
    // consume stage B
    c00 = __builtin_amdgcn_wmma_f32_16x16x32_bf16(false, a0B.v, false, b0B.v,
                                                  (short)0, c00, false, false);
    c10 = __builtin_amdgcn_wmma_f32_16x16x32_bf16(false, a1B.v, false, b0B.v,
                                                  (short)0, c10, false, false);
    c01 = __builtin_amdgcn_wmma_f32_16x16x32_bf16(false, a0B.v, false, b1B.v,
                                                  (short)0, c01, false, false);
    c11 = __builtin_amdgcn_wmma_f32_16x16x32_bf16(false, a1B.v, false, b1B.v,
                                                  (short)0, c11, false, false);
  }

  // C/D layout: VGPR r -> M = r + 8*half; column = lane&15
  const int col0 = n0 + lr;
  const int col1 = col0 + 16;
  const float bi0 = bias[col0];
  const float bi1 = bias[col1];
  #pragma unroll
  for (int r = 0; r < 8; ++r) {
    const int rowLo = m0 + r + half * 8;        // tile rows m0..m0+15
    const int rowHi = rowLo + 16;               // tile rows m0+16..m0+31
    float v00 = c00[r] + bi0;
    float v01 = c01[r] + bi1;
    float v10 = c10[r] + bi0;
    float v11 = c11[r] + bi1;
    if (RELU) {
      v00 = fmaxf(v00, 0.f); v01 = fmaxf(v01, 0.f);
      v10 = fmaxf(v10, 0.f); v11 = fmaxf(v11, 0.f);
    }
    if (OUT_BF16) {
      uint16_t* o = (uint16_t*)out;
      o[(size_t)rowLo * ldd + col0] = f2bf(v00);
      o[(size_t)rowLo * ldd + col1] = f2bf(v01);
      o[(size_t)rowHi * ldd + col0] = f2bf(v10);
      o[(size_t)rowHi * ldd + col1] = f2bf(v11);
    } else {
      float* o = (float*)out;
      o[(size_t)rowLo * ldd + col0] = v00;
      o[(size_t)rowLo * ldd + col1] = v01;
      o[(size_t)rowHi * ldd + col0] = v10;
      o[(size_t)rowHi * ldd + col1] = v11;
    }
  }
}

// ---------------------------------------------------------------------------
// Kernel 5: block reassembly + final 1x1 conv OC->CH, coalesced y write
// y[n,c,h,w] = sum_o R[n*16+b, o*256+s] * wout[c*7+o]
// ---------------------------------------------------------------------------
__global__ __launch_bounds__(256)
void k_final_conv(const float* __restrict__ R,
                  const float* __restrict__ wout,
                  float* __restrict__ y) {
  const int g = blockIdx.x * blockDim.x + threadIdx.x;
  const int w = g & 63;
  const int h = (g >> 6) & 63;
  const int c = (g >> 12) & 63;
  const int n = g >> 18;
  const int b = (h >> 4) * LRF_ + (w >> 4);
  const int s = (h & 15) * LH_ + (w & 15);
  const float* rp = R + (size_t)(n * B_ + b) * OUTDIM_ + s;
  float acc = 0.f;
  #pragma unroll
  for (int o = 0; o < OC_; ++o) acc += rp[o * S_] * wout[c * OC_ + o];
  y[g] = acc;
}

// ---------------------------------------------------------------------------
extern "C" void kernel_launch(void* const* d_in, const int* in_sizes, int n_in,
                              void* d_out, int out_size, void* d_ws, size_t ws_size,
                              hipStream_t stream) {
  (void)in_sizes; (void)n_in; (void)out_size; (void)ws_size;
  const float* x      = (const float*)d_in[0];
  const float* conv_w = (const float*)d_in[1];
  const float* W1     = (const float*)d_in[2];
  const float* b1     = (const float*)d_in[3];
  const float* W2     = (const float*)d_in[4];
  const float* b2     = (const float*)d_in[5];
  const float* wout   = (const float*)d_in[6];
  float* y = (float*)d_out;

  // Workspace layout (all offsets 256B-aligned):
  char* ws = (char*)d_ws;
  const size_t featB_bytes = (size_t)ROWS_ * INDIM_ * 2;       // 7,602,176
  const size_t w1t_bytes   = (size_t)HID_ * INDIM_ * 2;        // 6,889,472
  const size_t w2t_bytes   = (size_t)OUTDIM_ * HID_ * 2;       // 6,651,904
  const size_t hbf_bytes   = (size_t)ROWS_ * HID_ * 2;         // 7,602,176
  uint16_t* featB = (uint16_t*)(ws);
  uint16_t* W1t   = (uint16_t*)(ws + featB_bytes);
  uint16_t* W2t   = (uint16_t*)(ws + featB_bytes + w1t_bytes);
  uint16_t* Hbf   = (uint16_t*)(ws + featB_bytes + w1t_bytes + w2t_bytes);
  float*    Rf    = (float*)   (ws + featB_bytes + w1t_bytes + w2t_bytes + hbf_bytes);

  // 1) weight transpose/convert (fits in L2; done fresh each call = deterministic)
  k_transpose_f32_to_bf16<<<(INDIM_ * HID_   + 255) / 256, 256, 0, stream>>>(W1, W1t, INDIM_, HID_);
  k_transpose_f32_to_bf16<<<(HID_ * OUTDIM_  + 255) / 256, 256, 0, stream>>>(W2, W2t, HID_, OUTDIM_);

  // 2) softmax / top-k / gather -> bf16 feature rows [2048 x 1856]
  k_topk_gather<<<ROWS_, 256, 0, stream>>>(x, conv_w, featB);

  // 3) H = relu(feat @ W1 + b1)  [2048 x 1856], bf16 out
  k_gemm_wmma<true, true><<<dim3(HID_ / 64, ROWS_ / 128), 256, 0, stream>>>(
      featB, INDIM_, W1t, INDIM_, b1, (void*)Hbf, HID_);

  // 4) R = H @ W2 + b2  [2048 x 1792], f32 out
  k_gemm_wmma<false, false><<<dim3(OUTDIM_ / 64, ROWS_ / 128), 256, 0, stream>>>(
      Hbf, HID_, W2t, HID_, b2, (void*)Rf, OUTDIM_);

  // 5) reassemble + final 1x1 conv -> y [128,64,64,64]
  const int total = N_ * CH_ * RES_ * RES_;   // 33,554,432
  k_final_conv<<<total / 256, 256, 0, stream>>>(Rf, wout, y);
}
